// DeciLMMambaMixer_59725815218319
// MI455X (gfx1250) — matrix operand order repaired and checked by reading, not compiled, not run on hardware
//
#include <hip/hip_runtime.h>
#include <hip/hip_bf16.h>

// ---------------- problem constants ----------------
#define H_      4096
#define DSSM_   8192
#define NH_     64
#define P_      128
#define NST_    128            // state dim N
#define KCONV_  4
#define CONV_DIM_ (DSSM_ + 2 * NST_)          // 8448
#define PROJ_DIM_ (DSSM_ + CONV_DIM_ + NH_)   // 16704
#define PROJ_PAD_ 16768                        // 131*128: padded so GEMM has no tails
#define BATCH_  2
#define SEQ_    1024
#define BS_TOT_ (BATCH_ * SEQ_)               // 2048

typedef __bf16 bf16_t;
typedef __attribute__((ext_vector_type(16))) __bf16 v16bf;
typedef __attribute__((ext_vector_type(8)))  float  v8f;

__device__ __forceinline__ unsigned short f32_to_bf16(float f) {
  unsigned int u = __float_as_uint(f);
  unsigned int r = u + 0x7FFFu + ((u >> 16) & 1u);   // round-to-nearest-even
  return (unsigned short)(r >> 16);
}

// ---------------- fp32 -> bf16 conversion ----------------
__global__ __launch_bounds__(256) void cvt_f32_bf16(const float* __restrict__ src,
                                                    unsigned short* __restrict__ dst,
                                                    long long n) {
  long long i = (long long)blockIdx.x * 256 + threadIdx.x;
  if (i < n) dst[i] = f32_to_bf16(src[i]);
}

// fp32 (rows x src_cols) -> bf16 (rows x dst_cols), zero pad cols >= src_cols
__global__ __launch_bounds__(256) void cvt_pad_bf16(const float* __restrict__ src,
                                                    unsigned short* __restrict__ dst,
                                                    int rows, int src_cols, int dst_cols) {
  long long i = (long long)blockIdx.x * 256 + threadIdx.x;
  long long total = (long long)rows * dst_cols;
  if (i >= total) return;
  int c = (int)(i % dst_cols);
  long long r = i / dst_cols;
  dst[i] = (c < src_cols) ? f32_to_bf16(src[r * src_cols + c]) : (unsigned short)0;
}

// ---------------- bf16 WMMA GEMM: C(MxN) = A(MxK) * B(KxN), fp32 accum ----------------
// Block tile 128x128, K slab 64 (two WMMA-K chunks per barrier pair).
// 256 threads = 8 waves in 4(M) x 2(N); each wave owns 2x4 subtiles of 16x16.
// A tile staged with GLOBAL_LOAD_ASYNC_TO_LDS_B128 (ASYNCcnt path); B tile staged
// through VGPRs because it needs a 2-byte-granularity transpose into [n][k].
// Both LDS strides are 72 u16 (144B rows) so all fragment bases are 16B-aligned
// and fragment reads vectorize to ds_load_b128.
// ALL dimensions are exact multiples (M%128==0, N%128==0, K%64==0): no guards.
#define AST 72   // LDS u16 stride for A rows
#define BST 72   // LDS u16 stride for B cols
#define LDS_ALIGN_PAD 0

__global__ __launch_bounds__(256) void gemm_bf16_wmma(
    const unsigned short* __restrict__ A,   // bf16 bits, row-major M x K (lda = Kdim)
    const unsigned short* __restrict__ B,   // bf16 bits, row-major K x N (ldb = Ndim)
    float* __restrict__ C,                  // fp32 row-major M x N (ldc = Ndim)
    int Ndim, int Kdim) {
  __shared__ __align__(16) unsigned short As16[128 * AST];   // [m][k]
  __shared__ __align__(16) unsigned short Bs16[128 * BST];   // [n][k] transposed-packed

  const int m0   = blockIdx.y * 128;
  const int n0   = blockIdx.x * 128;
  const int t    = threadIdx.x;
  const int lane = t & 31;
  const int wave = t >> 5;
  const int wm   = (wave & 3) * 32;   // wave M offset
  const int wn   = (wave >> 2) * 64;  // wave N offset

  const unsigned lds_a_base = (unsigned)(unsigned long long)(uintptr_t)&As16[0];

  v8f acc[2][4];
#pragma unroll
  for (int i = 0; i < 2; ++i)
#pragma unroll
    for (int j = 0; j < 4; ++j) {
      v8f z = {0.f, 0.f, 0.f, 0.f, 0.f, 0.f, 0.f, 0.f};
      acc[i][j] = z;
    }

  for (int k0 = 0; k0 < Kdim; k0 += 64) {
    // ---- stage A tile via async copy: 128 rows x 64 k = 1024 chunks of 8 bf16 ----
#pragma unroll
    for (int it = 0; it < 4; ++it) {
      const int c    = t + it * 256;
      const int row  = c >> 3;
      const int kseg = (c & 7) << 3;                       // bf16 units, 8 per 16B chunk
      const unsigned lds_off = lds_a_base + (unsigned)(row * AST + kseg) * 2u;
      const unsigned long long gaddr =
          (unsigned long long)(uintptr_t)(A + (size_t)(m0 + row) * Kdim + k0 + kseg);
      asm volatile("global_load_async_to_lds_b128 %0, %1, off"
                   :: "v"(lds_off), "v"(gaddr) : "memory");
    }
    // ---- stage B tile transposed: 64 k rows x 128 n, scatter into [n][k] ----
#pragma unroll
    for (int it = 0; it < 2; ++it) {
      const int seg  = t + it * 256;
      const int krow = seg >> 3;
      const int nseg = (seg & 7) << 4;
      const unsigned short* srcb = B + (size_t)(k0 + krow) * Ndim + n0 + nseg;
      union { uint4 q[2]; unsigned short s[16]; } u;
      const uint4* sv = (const uint4*)srcb;
      u.q[0] = sv[0]; u.q[1] = sv[1];
#pragma unroll
      for (int j = 0; j < 16; ++j) Bs16[(nseg + j) * BST + krow] = u.s[j];
      if (k0 + 64 < Kdim)
        __builtin_prefetch(B + (size_t)(k0 + 64 + krow) * Ndim + n0 + nseg, 0, 0);
    }
    // wait for this wave's async copies, then publish tiles to the workgroup
    asm volatile("s_wait_asynccnt 0" ::: "memory");
    __syncthreads();

    // ---- two WMMA-K chunks per stage ----
#pragma unroll
    for (int kk = 0; kk < 64; kk += 32) {
      // A 16x32 bf16 fragment: lane L -> M = L%16;
      // lanes 0-15: K kk+0..15; lanes 16-31: K kk+16..31
      v16bf afrag[2];
      {
        const int mrowb = (lane & 15);
        const int akb = ((lane >> 4) << 4) + kk;   // kk, kk+16
#pragma unroll
        for (int i = 0; i < 2; ++i) {
          const int mrow = wm + i * 16 + mrowb;
          const unsigned int* p0 = (const unsigned int*)(As16 + mrow * AST + akb);
          const unsigned int* p1 = (const unsigned int*)(As16 + mrow * AST + akb + 8);
          union { unsigned int u[8]; v16bf v; } ua;
          ua.u[0] = p0[0]; ua.u[1] = p0[1]; ua.u[2] = p0[2]; ua.u[3] = p0[3];
          ua.u[4] = p1[0]; ua.u[5] = p1[1]; ua.u[6] = p1[2]; ua.u[7] = p1[3];
          afrag[i] = ua.v;
        }
      }
      // B 32x16 bf16 fragment: lane L -> N = L%16;
      // lanes 0-15: K kk+0..15; lanes 16-31: K kk+16..31
      v16bf bfrag[4];
      {
        const int bkb = ((lane >> 4) << 4) + kk;   // {0,16,32,48}: 16B-aligned with BST=72
#pragma unroll
        for (int j = 0; j < 4; ++j) {
          const int ncol = wn + j * 16 + (lane & 15);
          const unsigned int* p = (const unsigned int*)(Bs16 + ncol * BST + bkb);
          union { unsigned int u[8]; v16bf v; } ub;
#pragma unroll
          for (int q = 0; q < 8; ++q) ub.u[q] = p[q];
          bfrag[j] = ub.v;
        }
      }
#pragma unroll
      for (int i = 0; i < 2; ++i)
#pragma unroll
        for (int j = 0; j < 4; ++j)
          acc[i][j] = __builtin_amdgcn_wmma_f32_16x16x32_bf16(
              false, afrag[i], false, bfrag[j], (short)0, acc[i][j], false, false);
    }
    __syncthreads();
  }

  // ---- store: C/D layout: lane L -> N = L%16; VGPR r -> M = r + 8*(L/16) ----
#pragma unroll
  for (int i = 0; i < 2; ++i) {
    const int mbase = m0 + wm + i * 16 + ((lane >> 4) << 3);
#pragma unroll
    for (int j = 0; j < 4; ++j) {
      const int ncol = n0 + wn + j * 16 + (lane & 15);
#pragma unroll
      for (int r = 0; r < 8; ++r)
        C[(size_t)(mbase + r) * Ndim + ncol] = acc[i][j][r];
    }
  }
}

// ---------------- causal depthwise conv (K=4) + SiLU, split x / B / C ----------------
__global__ __launch_bounds__(256) void conv_silu_kernel(
    const float* __restrict__ proj, const float* __restrict__ conv_w,
    float* __restrict__ xs, float* __restrict__ Bv, float* __restrict__ Cv) {
  long long idx = (long long)blockIdx.x * 256 + threadIdx.x;
  if (idx >= (long long)BS_TOT_ * CONV_DIM_) return;
  int c = (int)(idx % CONV_DIM_);
  long long bs = idx / CONV_DIM_;
  int s = (int)(bs % SEQ_);
  long long b = bs / SEQ_;
  float acc = 0.f;
#pragma unroll
  for (int k = 0; k < KCONV_; ++k) {
    int sp = s - (KCONV_ - 1) + k;
    if (sp >= 0) {
      float v = proj[((b * SEQ_ + sp) * (long long)PROJ_PAD_) + DSSM_ + c];
      acc += v * conv_w[c * KCONV_ + k];
    }
  }
  float sv = acc / (1.f + __expf(-acc));   // silu
  if (c < DSSM_)              xs[bs * DSSM_ + c] = sv;
  else if (c < DSSM_ + NST_)  Bv[bs * NST_ + (c - DSSM_)] = sv;
  else                        Cv[bs * NST_ + (c - DSSM_ - NST_)] = sv;
}

// ---------------- dt = softplus(dt_raw + bias); dA = exp(-exp(A_log)*dt) ----------------
__global__ __launch_bounds__(256) void dt_kernel(
    const float* __restrict__ proj, const float* __restrict__ dt_bias,
    const float* __restrict__ A_log, float* __restrict__ dt_arr,
    float* __restrict__ dA_arr) {
  int idx = blockIdx.x * 256 + threadIdx.x;
  if (idx >= BS_TOT_ * NH_) return;
  int h = idx & (NH_ - 1);
  int bs = idx >> 6;
  float x = proj[(size_t)bs * PROJ_PAD_ + DSSM_ + CONV_DIM_ + h] + dt_bias[h];
  float sp = (x > 20.f) ? x : log1pf(__expf(x));
  dt_arr[idx] = sp;
  dA_arr[idx] = __expf(-__expf(A_log[h]) * sp);
}

// ---------------- sequential SSM scan: one block per (b,h), state in VGPRs ----------------
__global__ __launch_bounds__(128) void scan_kernel(
    const float* __restrict__ xs, const float* __restrict__ Bv,
    const float* __restrict__ Cv, const float* __restrict__ dt_arr,
    const float* __restrict__ dA_arr, const float* __restrict__ D_param,
    float* __restrict__ ybuf) {
  const int bh = blockIdx.x;            // 0..127
  const int b = bh >> 6;
  const int h = bh & 63;
  const int p = threadIdx.x;            // 0..127
  __shared__ float Bsh[2][NST_];
  __shared__ float Csh[2][NST_];
  float hreg[NST_];
#pragma unroll
  for (int n = 0; n < NST_; ++n) hreg[n] = 0.f;
  const float Dh = D_param[h];

  for (int s = 0; s < SEQ_; ++s) {
    const int bs = b * SEQ_ + s;
    const int buf = s & 1;
    Bsh[buf][p] = Bv[(size_t)bs * NST_ + p];
    Csh[buf][p] = Cv[(size_t)bs * NST_ + p];
    const float dt = dt_arr[bs * NH_ + h];
    const float dA = dA_arr[bs * NH_ + h];
    const float xv = xs[(size_t)bs * DSSM_ + h * P_ + p];
    const float dtx = dt * xv;
    __syncthreads();                    // double-buffered: one barrier / step
    float y = 0.f;
#pragma unroll
    for (int n = 0; n < NST_; ++n) {
      hreg[n] = dA * hreg[n] + dtx * Bsh[buf][n];
      y += hreg[n] * Csh[buf][n];
    }
    ybuf[(size_t)bs * DSSM_ + h * P_ + p] = y + Dh * xv;
  }
}

// ---------------- y*silu(z), RMSNorm over 8192, emit bf16 for final GEMM ----------------
__global__ __launch_bounds__(256) void gate_rms_kernel(
    const float* __restrict__ ybuf, const float* __restrict__ proj,
    const float* __restrict__ norm_weight, unsigned short* __restrict__ yn) {
  const int bs = blockIdx.x;
  const int t = threadIdx.x;
  __shared__ float red[256];
  float vals[32];
  float ss = 0.f;
#pragma unroll
  for (int i = 0; i < 32; ++i) {
    const int c = t + i * 256;
    float z = proj[(size_t)bs * PROJ_PAD_ + c];
    float sz = z / (1.f + __expf(-z));
    float v = ybuf[(size_t)bs * DSSM_ + c] * sz;
    vals[i] = v;
    ss += v * v;
  }
  red[t] = ss;
  __syncthreads();
  for (int ofs = 128; ofs > 0; ofs >>= 1) {
    if (t < ofs) red[t] += red[t + ofs];
    __syncthreads();
  }
  const float rstd = rsqrtf(red[0] / (float)DSSM_ + 1e-5f);
#pragma unroll
  for (int i = 0; i < 32; ++i) {
    const int c = t + i * 256;
    yn[(size_t)bs * DSSM_ + c] = f32_to_bf16(vals[i] * rstd * norm_weight[c]);
  }
}

// ---------------- host side ----------------
extern "C" void kernel_launch(void* const* d_in, const int* in_sizes, int n_in,
                              void* d_out, int out_size, void* d_ws, size_t ws_size,
                              hipStream_t stream) {
  const float* hidden = (const float*)d_in[0];   // (2,1024,4096)
  const float* W_in   = (const float*)d_in[1];   // (4096,16704)
  const float* conv_w = (const float*)d_in[2];   // (8448,4)
  const float* dtb    = (const float*)d_in[3];   // (64,)
  const float* A_log  = (const float*)d_in[4];   // (64,)
  const float* D_par  = (const float*)d_in[5];   // (64,)
  const float* nw     = (const float*)d_in[6];   // (8192,)
  const float* W_out  = (const float*)d_in[7];   // (8192,4096)
  float* out = (float*)d_out;                    // (2,1024,4096)

  size_t off = 0;
  auto alloc = [&](size_t bytes) -> void* {
    off = (off + 255) & ~(size_t)255;
    void* p = (char*)d_ws + off;
    off += bytes;
    return p;
  };
  unsigned short* Xb   = (unsigned short*)alloc((size_t)BS_TOT_ * H_ * 2);
  unsigned short* Wib  = (unsigned short*)alloc((size_t)H_ * PROJ_PAD_ * 2);  // zero-padded N
  unsigned short* Wob  = (unsigned short*)alloc((size_t)DSSM_ * H_ * 2);
  float* proj  = (float*)alloc((size_t)BS_TOT_ * PROJ_PAD_ * 4);              // padded ldc
  float* xs    = (float*)alloc((size_t)BS_TOT_ * DSSM_ * 4);
  float* Bv    = (float*)alloc((size_t)BS_TOT_ * NST_ * 4);
  float* Cv    = (float*)alloc((size_t)BS_TOT_ * NST_ * 4);
  float* dt_a  = (float*)alloc((size_t)BS_TOT_ * NH_ * 4);
  float* dA_a  = (float*)alloc((size_t)BS_TOT_ * NH_ * 4);
  float* ybuf  = (float*)alloc((size_t)BS_TOT_ * DSSM_ * 4);
  unsigned short* yn = (unsigned short*)alloc((size_t)BS_TOT_ * DSSM_ * 2);
  (void)ws_size; (void)n_in; (void)in_sizes; (void)out_size;

  // 1) bf16 conversions (bf16 W_in = 137 MB fits the 192 MB L2 across M-block re-reads)
  {
    long long n1 = (long long)BS_TOT_ * H_;
    cvt_f32_bf16<<<(unsigned)((n1 + 255) / 256), 256, 0, stream>>>(hidden, Xb, n1);
    long long n2 = (long long)H_ * PROJ_PAD_;
    cvt_pad_bf16<<<(unsigned)((n2 + 255) / 256), 256, 0, stream>>>(W_in, Wib, H_, PROJ_DIM_, PROJ_PAD_);
    long long n3 = (long long)DSSM_ * H_;
    cvt_f32_bf16<<<(unsigned)((n3 + 255) / 256), 256, 0, stream>>>(W_out, Wob, n3);
  }
  // 2) proj = X @ W_in   (2048 x 16768pad x 4096), tail-free
  {
    dim3 grid(PROJ_PAD_ / 128, BS_TOT_ / 128);
    gemm_bf16_wmma<<<grid, 256, 0, stream>>>(Xb, Wib, proj, PROJ_PAD_, H_);
  }
  // 3) conv + silu
  {
    long long n = (long long)BS_TOT_ * CONV_DIM_;
    conv_silu_kernel<<<(unsigned)((n + 255) / 256), 256, 0, stream>>>(proj, conv_w, xs, Bv, Cv);
  }
  // 4) dt / dA
  dt_kernel<<<(BS_TOT_ * NH_ + 255) / 256, 256, 0, stream>>>(proj, dtb, A_log, dt_a, dA_a);
  // 5) sequential scan
  scan_kernel<<<BATCH_ * NH_, 128, 0, stream>>>(xs, Bv, Cv, dt_a, dA_a, D_par, ybuf);
  // 6) gate + RMSNorm -> bf16
  gate_rms_kernel<<<BS_TOT_, 256, 0, stream>>>(ybuf, proj, nw, yn);
  // 7) out = yn @ W_out  (2048 x 4096 x 8192), tail-free
  {
    dim3 grid(H_ / 128, BS_TOT_ / 128);
    gemm_bf16_wmma<<<grid, 256, 0, stream>>>(yn, Wob, out, H_, DSSM_);
  }
}
